// UniMatrixBlock_69123203662013
// MI455X (gfx1250) — matrix operand order repaired and checked by reading, not compiled, not run on hardware
//
#include <hip/hip_runtime.h>
#include <hip/hip_bf16.h>

// ---------------------------------------------------------------------------
// Problem constants (from the reference)
// ---------------------------------------------------------------------------
#define Bc 4
#define Tc 512
#define Dc 1024
#define Hc 16
#define Sc 64
#define INNERc (Hc * Sc)   // 1024
#define HIDc (4 * Dc)      // 4096
#define Mc (Bc * Tc)       // 2048 rows in every GEMM

// ---------------------------------------------------------------------------
// bf16 helpers (no __bf16 arithmetic; pure bit manipulation + bit_cast)
// ---------------------------------------------------------------------------
typedef __attribute__((ext_vector_type(16))) __bf16          v16bf;
typedef __attribute__((ext_vector_type(16))) unsigned short  v16u;
typedef __attribute__((ext_vector_type(8)))  unsigned short  v8us;
typedef __attribute__((ext_vector_type(8)))  float           v8f;

static __device__ __forceinline__ unsigned short f2bf(float f) {
    unsigned u = __builtin_bit_cast(unsigned, f);
    unsigned r = u + 0x7FFFu + ((u >> 16) & 1u);   // round-to-nearest-even
    return (unsigned short)(r >> 16);
}
static __device__ __forceinline__ void storeOut(float* p, float v) { *p = v; }
static __device__ __forceinline__ void storeOut(unsigned short* p, float v) { *p = f2bf(v); }

static __device__ __forceinline__ v16u loadAfrag(const unsigned short* Arow, int k0) {
    const v8us lo = *(const v8us*)(Arow + k0);
    const v8us hi = *(const v8us*)(Arow + k0 + 16);
    v16u r;
#pragma unroll
    for (int p = 0; p < 8; ++p) { r[p] = lo[p]; r[8 + p] = hi[p]; }
    return r;
}
static __device__ __forceinline__ v8f wmma_bf16(const v16u& a, const v16u& b, const v8f& c) {
    return __builtin_amdgcn_wmma_f32_16x16x32_bf16(
        false, __builtin_bit_cast(v16bf, a),
        false, __builtin_bit_cast(v16bf, b),
        (short)0, c, false, false);
}

// ---------------------------------------------------------------------------
// Transpose-cast: src f32 [K,N] -> dst bf16 [N,K]  (LDS tiled, coalesced both
// ways). Run once per launch over the weights so every WMMA B-fragment becomes
// two 16-byte loads of 8 contiguous k each.
// ---------------------------------------------------------------------------
__global__ __launch_bounds__(256) void transpose_cast_kernel(const float* __restrict__ src,
                                                             unsigned short* __restrict__ dst,
                                                             int K, int N) {
    __shared__ unsigned short tile[32][33];
    const int kt = blockIdx.y * 32, nt = blockIdx.x * 32;
    const int tx = threadIdx.x & 31, ty = threadIdx.x >> 5;   // 32 x 8
#pragma unroll
    for (int s = 0; s < 32; s += 8) {
        const int k = kt + ty + s, n = nt + tx;
        tile[ty + s][tx] = (k < K && n < N) ? f2bf(src[(size_t)k * N + n]) : (unsigned short)0;
    }
    __syncthreads();
#pragma unroll
    for (int s = 0; s < 32; s += 8) {
        const int n = nt + ty + s, k = kt + tx;
        if (n < N && k < K) dst[(size_t)n * K + k] = tile[tx][ty + s];
    }
}

// ---------------------------------------------------------------------------
// Row LayerNorm: [rows, D] f32 -> bf16 normalized*w+b
// ---------------------------------------------------------------------------
__global__ __launch_bounds__(256) void ln_bf16_kernel(const float* __restrict__ x,
                                                      const float* __restrict__ w,
                                                      const float* __restrict__ b,
                                                      unsigned short* __restrict__ out) {
    const int row = blockIdx.x;
    const float* xr = x + (size_t)row * Dc;
    __shared__ float s1[256], s2[256], stat[2];
    float a0 = 0.f, a1 = 0.f;
    for (int i = threadIdx.x; i < Dc; i += 256) { float v = xr[i]; a0 += v; a1 += v * v; }
    s1[threadIdx.x] = a0; s2[threadIdx.x] = a1;
    __syncthreads();
    for (int off = 128; off > 0; off >>= 1) {
        if (threadIdx.x < off) { s1[threadIdx.x] += s1[threadIdx.x + off]; s2[threadIdx.x] += s2[threadIdx.x + off]; }
        __syncthreads();
    }
    if (threadIdx.x == 0) {
        float mean = s1[0] * (1.f / Dc);
        float var  = s2[0] * (1.f / Dc) - mean * mean;
        stat[0] = mean; stat[1] = rsqrtf(var + 1e-5f);
    }
    __syncthreads();
    const float mean = stat[0], rstd = stat[1];
    unsigned short* o = out + (size_t)row * Dc;
    for (int i = threadIdx.x; i < Dc; i += 256)
        o[i] = f2bf((xr[i] - mean) * rstd * w[i] + b[i]);
}

// ---------------------------------------------------------------------------
// bf16 WMMA GEMM with pre-transposed B:
//   C[M,N] = epilogue(A[M,K] @ Bt[N,K]^T + bias [+ residual])
// Register-blocked: each wave computes a 32x32 tile (2x2 of 16x16), 8 waves
// per block -> 64(M) x 128(N) block tile. Double-buffered fragments: next
// K-step's 8x global_load_b128 are issued before the current 4 WMMAs so the
// loads execute underneath the matrix pipeline. 2 b128 loads per WMMA.
// ACT: 0=none 1=tanh 2=sigmoid 3=gelu(exact)
// ---------------------------------------------------------------------------
template <int ACT, typename OutT>
__global__ __launch_bounds__(256) void gemm_wmma_bf16(const unsigned short* __restrict__ A,
                                                      const unsigned short* __restrict__ Bt,
                                                      const float* __restrict__ bias,
                                                      const float* __restrict__ res,
                                                      OutT* __restrict__ C,
                                                      int M, int N, int K) {
    const int wave = threadIdx.x >> 5;
    const int lane = threadIdx.x & 31;
    const int m0 = blockIdx.y * 64 + (wave >> 2) * 32;
    const int n0 = blockIdx.x * 128 + (wave & 3) * 32;
    if (m0 >= M || n0 >= N) return;

    const int row  = lane & 15;         // M-row for A frags; N-column for B/C frags
    const int half = lane >> 4;
    const bool n1ok = (n0 + 16) < N;    // second N-subtile in range? (N=16/48 cases)

    // A frag per lane: K = k0 + half*8 + {0..7}  and  k0 + 16 + half*8 + {0..7}
    const unsigned short* Arow0 = A + (size_t)(m0 + row) * K + half * 8;
    const unsigned short* Arow1 = Arow0 + (size_t)16 * K;
    // B frag per lane: K = k0 + half*8 + {0..7}  and  k0 + 16 + half*8 + {0..7} at fixed output column
    const unsigned short* Bcol0 = Bt + (size_t)(n0 + row) * K + half * 8;
    const unsigned short* Bcol1 = n1ok ? (Bcol0 + (size_t)16 * K) : Bcol0;  // alias if OOB

    v8f acc00 = {}, acc01 = {}, acc10 = {}, acc11 = {};
    v16u a0 = loadAfrag(Arow0, 0);
    v16u a1 = loadAfrag(Arow1, 0);
    v16u b0 = loadAfrag(Bcol0, 0);
    v16u b1 = loadAfrag(Bcol1, 0);

    for (int k0 = 0; k0 < K; k0 += 32) {
        v16u a0n = a0, a1n = a1, b0n = b0, b1n = b1;
        const bool more = (k0 + 32) < K;
        if (more) {
            if (k0 + 288 < K) {                               // gfx1250 global_prefetch_b8
                __builtin_prefetch((const void*)(Arow0 + k0 + 288), 0, 3);
                __builtin_prefetch((const void*)(Arow1 + k0 + 288), 0, 3);
                __builtin_prefetch((const void*)(Bcol0 + k0 + 288), 0, 3);
                __builtin_prefetch((const void*)(Bcol1 + k0 + 288), 0, 3);
            }
            a0n = loadAfrag(Arow0, k0 + 32);
            a1n = loadAfrag(Arow1, k0 + 32);
            b0n = loadAfrag(Bcol0, k0 + 32);
            b1n = loadAfrag(Bcol1, k0 + 32);
        }
        acc00 = wmma_bf16(a0, b0, acc00);
        acc01 = wmma_bf16(a0, b1, acc01);
        acc10 = wmma_bf16(a1, b0, acc10);
        acc11 = wmma_bf16(a1, b1, acc11);
        a0 = a0n; a1 = a1n; b0 = b0n; b1 = b1n;
    }

    // ---- epilogue: 4 sub-tiles ----
    const int colA = n0 + row, colB = n0 + 16 + row;
    auto emit = [&](const v8f& acc, int mb, int cc, bool ok) {
        if (!ok) return;
        const float bv = bias ? bias[cc] : 0.f;
#pragma unroll
        for (int v = 0; v < 8; ++v) {
            const int r = mb + v + half * 8;
            float val = acc[v] + bv;
            if (ACT == 1) val = tanhf(val);
            else if (ACT == 2) val = 1.f / (1.f + expf(-val));
            else if (ACT == 3) val = 0.5f * val * (1.f + erff(val * 0.70710678118654752f));
            if (res) val += res[(size_t)r * N + cc];
            storeOut(&C[(size_t)r * N + cc], val);
        }
    };
    emit(acc00, m0,      colA, true);
    emit(acc01, m0,      colB, n1ok);
    emit(acc10, m0 + 16, colA, true);
    emit(acc11, m0 + 16, colB, n1ok);
}

// ---------------------------------------------------------------------------
// 3-way softmax over rule logits [B*T*H, 3], in place
// ---------------------------------------------------------------------------
__global__ __launch_bounds__(256) void softmax3_kernel(float* __restrict__ p, int rows) {
    int i = blockIdx.x * blockDim.x + threadIdx.x;
    if (i >= rows) return;
    float a = p[i * 3 + 0], b = p[i * 3 + 1], c = p[i * 3 + 2];
    float m = fmaxf(a, fmaxf(b, c));
    float ea = expf(a - m), eb = expf(b - m), ec = expf(c - m);
    float inv = 1.f / (ea + eb + ec);
    p[i * 3 + 0] = ea * inv; p[i * 3 + 1] = eb * inv; p[i * 3 + 2] = ec * inv;
}

// ---------------------------------------------------------------------------
// Sequential fast-weight scan: one block per (b,h); 64x64 state in LDS.
// q,k,v,dg come fused in one [M, 4*INNER] buffer (column blocks).
// Writes per-token readout as bf16 (feeds Wo GEMM) and final state (f32).
// ---------------------------------------------------------------------------
__global__ __launch_bounds__(256) void scan_kernel(const float* __restrict__ qkvd,
                                                   const float* __restrict__ ret,
                                                   const float* __restrict__ rw,
                                                   const float* __restrict__ st_in,
                                                   unsigned short* __restrict__ out_bf,
                                                   float* __restrict__ state_out) {
    const int bh = blockIdx.x;
    const int b = bh / Hc, h = bh % Hc;
    const int tid = threadIdx.x;

    __shared__ float st[Sc][Sc + 1];   // padded: stride 65 avoids bank conflicts
    __shared__ float red[256];
    __shared__ float vec[4 * Sc];      // q | k | v | dg
    __shared__ float scal[8];          // r, w0, w1, w2, scale

    const float* sti = st_in + (size_t)bh * Sc * Sc;
    for (int i = tid; i < Sc * Sc; i += 256) st[i >> 6][i & 63] = sti[i];
    __syncthreads();

    for (int t = 0; t < Tc; ++t) {
        const size_t rowb = (size_t)(b * Tc + t);
        const size_t qb   = rowb * (4 * INNERc) + h * Sc;     // q at col offset 0
        const size_t base = (rowb * Hc + h) * Sc;             // out_bf index
        if (tid < Sc) {
            vec[tid]          = qkvd[qb + tid];                 // q
            vec[Sc + tid]     = qkvd[qb + INNERc + tid];        // k
            vec[2 * Sc + tid] = qkvd[qb + 2 * INNERc + tid];    // v
            vec[3 * Sc + tid] = qkvd[qb + 3 * INNERc + tid];    // dg
        }
        if (tid == 0) {
            const size_t rb = rowb * Hc + h;
            scal[0] = ret[rb];
            scal[1] = rw[rb * 3 + 0];
            scal[2] = rw[rb * 3 + 1];
            scal[3] = rw[rb * 3 + 2];
        }
        __syncthreads();
        const float r = scal[0], w0 = scal[1], w1 = scal[2], w2 = scal[3];

        float local = 0.f;
        for (int i = tid; i < Sc * Sc; i += 256) {
            const int d = i >> 6, e = i & 63;
            const float o   = vec[Sc + d] * vec[2 * Sc + e];   // k[d]*v[e]
            const float ot  = vec[Sc + e] * vec[2 * Sc + d];   // k[e]*v[d]
            const float dia = (d == e) ? vec[3 * Sc + d] : 0.f;
            const float s = r * st[d][e] + w0 * o + w1 * dia + w2 * 0.5f * (o + ot);
            st[d][e] = s;
            local += s * s;
        }
        red[tid] = local;
        __syncthreads();
        for (int off = 128; off > 0; off >>= 1) {
            if (tid < off) red[tid] += red[tid + off];
            __syncthreads();
        }
        if (tid == 0) scal[4] = 1.f / fmaxf(sqrtf(red[0]), 1.f);
        __syncthreads();
        const float scale = scal[4];
        for (int i = tid; i < Sc * Sc; i += 256) st[i >> 6][i & 63] *= scale;
        __syncthreads();

        // readout: out[e] = sum_d q[d] * st[d][e]; 4 partial sums per column
        const int e = tid & 63, part = tid >> 6;
        float p = 0.f;
        const int d0 = part * 16;
        for (int d = d0; d < d0 + 16; ++d) p += vec[d] * st[d][e];
        red[tid] = p;
        __syncthreads();
        if (part == 0)
            out_bf[base + e] = f2bf(red[e] + red[64 + e] + red[128 + e] + red[192 + e]);
        __syncthreads();
    }

    float* sto = state_out + (size_t)bh * Sc * Sc;
    for (int i = tid; i < Sc * Sc; i += 256) sto[i] = st[i >> 6][i & 63];
}

// ---------------------------------------------------------------------------
// h = x + attn * gate
// ---------------------------------------------------------------------------
__global__ __launch_bounds__(256) void combine_kernel(const float* __restrict__ x,
                                                      const float* __restrict__ attn,
                                                      const float* __restrict__ gate,
                                                      float* __restrict__ h, int n) {
    for (int i = blockIdx.x * blockDim.x + threadIdx.x; i < n; i += gridDim.x * blockDim.x)
        h[i] = x[i] + attn[i] * gate[i];
}

// ---------------------------------------------------------------------------
// Host orchestration
// ---------------------------------------------------------------------------
extern "C" void kernel_launch(void* const* d_in, const int* in_sizes, int n_in,
                              void* d_out, int out_size, void* d_ws, size_t ws_size,
                              hipStream_t stream) {
    const float* x      = (const float*)d_in[0];
    // d_in[1] token_ids: unused by the reference
    const float* state0 = (const float*)d_in[2];
    const float* norm_w = (const float*)d_in[3];
    const float* norm_b = (const float*)d_in[4];
    const float* fnw    = (const float*)d_in[5];
    const float* fnb    = (const float*)d_in[6];
    const float* Wq     = (const float*)d_in[7];
    const float* Wk     = (const float*)d_in[8];
    const float* Wv     = (const float*)d_in[9];
    const float* Wd     = (const float*)d_in[10];
    const float* Wr     = (const float*)d_in[11];
    const float* br     = (const float*)d_in[12];
    const float* Wrule  = (const float*)d_in[13];
    const float* brule  = (const float*)d_in[14];
    const float* Wo     = (const float*)d_in[15];
    const float* Wg     = (const float*)d_in[16];
    const float* bg     = (const float*)d_in[17];
    const float* W1     = (const float*)d_in[18];
    const float* b1     = (const float*)d_in[19];
    const float* W2     = (const float*)d_in[20];
    const float* b2     = (const float*)d_in[21];

    float* out_f   = (float*)d_out;                       // h + ffn  [B,T,D]
    float* state_f = out_f + (size_t)Mc * Dc;             // state_f  [B,H,S,S]

    // bump allocator over scratch
    size_t off = 0;
    auto alloc = [&](size_t bytes) -> char* {
        char* p = (char*)d_ws + off;
        off += (bytes + 255) & ~(size_t)255;
        return p;
    };
    typedef unsigned short bfp;
    bfp* nrm_bf    = (bfp*)alloc((size_t)Mc * Dc * 2);
    bfp* Wqkvd_t   = (bfp*)alloc((size_t)4 * INNERc * Dc * 2);   // [4096,1024] = Wq^T|Wk^T|Wv^T|Wd^T
    bfp* Wr_t      = (bfp*)alloc((size_t)Hc * Dc * 2);           // [16,1024]
    bfp* Wrl_t     = (bfp*)alloc((size_t)3 * Hc * Dc * 2);       // [48,1024]
    bfp* Wo_t      = (bfp*)alloc((size_t)Dc * INNERc * 2);       // [1024,1024]
    bfp* Wg_t      = (bfp*)alloc((size_t)Dc * Dc * 2);           // [1024,1024]
    bfp* W1_t      = (bfp*)alloc((size_t)HIDc * Dc * 2);         // [4096,1024]
    bfp* W2_t      = (bfp*)alloc((size_t)Dc * HIDc * 2);         // [1024,4096]
    float* qkvd    = (float*)alloc((size_t)Mc * 4 * INNERc * 4); // tanh proj, fused
    float* retb    = (float*)alloc((size_t)Mc * Hc * 4);
    float* rwb     = (float*)alloc((size_t)Mc * 3 * Hc * 4);
    float* gateb   = (float*)alloc((size_t)Mc * Dc * 4);
    bfp* out_bf    = (bfp*)alloc((size_t)Mc * INNERc * 2);
    float* attnb   = (float*)alloc((size_t)Mc * Dc * 4);
    float* hb      = (float*)alloc((size_t)Mc * Dc * 4);
    bfp* h2_bf     = (bfp*)alloc((size_t)Mc * Dc * 2);
    bfp* ffn1_bf   = (bfp*)alloc((size_t)Mc * HIDc * 2);
    (void)ws_size; (void)in_sizes; (void)n_in; (void)out_size;

    const dim3 blk(256);
    auto tgrid = [](int K, int N) { return dim3((N + 31) / 32, (K + 31) / 32); };

    // 1) weights -> bf16, transposed to [N,K]
    transpose_cast_kernel<<<tgrid(Dc, INNERc), blk, 0, stream>>>(Wq, Wqkvd_t + (size_t)0 * INNERc * Dc, Dc, INNERc);
    transpose_cast_kernel<<<tgrid(Dc, INNERc), blk, 0, stream>>>(Wk, Wqkvd_t + (size_t)1 * INNERc * Dc, Dc, INNERc);
    transpose_cast_kernel<<<tgrid(Dc, INNERc), blk, 0, stream>>>(Wv, Wqkvd_t + (size_t)2 * INNERc * Dc, Dc, INNERc);
    transpose_cast_kernel<<<tgrid(Dc, INNERc), blk, 0, stream>>>(Wd, Wqkvd_t + (size_t)3 * INNERc * Dc, Dc, INNERc);
    transpose_cast_kernel<<<tgrid(Dc, Hc),     blk, 0, stream>>>(Wr,    Wr_t,  Dc, Hc);
    transpose_cast_kernel<<<tgrid(Dc, 3 * Hc), blk, 0, stream>>>(Wrule, Wrl_t, Dc, 3 * Hc);
    transpose_cast_kernel<<<tgrid(INNERc, Dc), blk, 0, stream>>>(Wo, Wo_t, INNERc, Dc);
    transpose_cast_kernel<<<tgrid(Dc, Dc),     blk, 0, stream>>>(Wg, Wg_t, Dc, Dc);
    transpose_cast_kernel<<<tgrid(Dc, HIDc),   blk, 0, stream>>>(W1, W1_t, Dc, HIDc);
    transpose_cast_kernel<<<tgrid(HIDc, Dc),   blk, 0, stream>>>(W2, W2_t, HIDc, Dc);

    // 2) LayerNorm -> bf16 activations
    ln_bf16_kernel<<<Mc, 256, 0, stream>>>(x, norm_w, norm_b, nrm_bf);

    auto grid_for = [](int N) { return dim3((N + 127) / 128, Mc / 64); };

    // 3) projections (WMMA, fused activations); q|k|v|dg fused: N = 4*INNER
    gemm_wmma_bf16<1, float><<<grid_for(4 * INNERc), blk, 0, stream>>>(nrm_bf, Wqkvd_t, nullptr, nullptr, qkvd, Mc, 4 * INNERc, Dc);
    gemm_wmma_bf16<2, float><<<grid_for(Hc),         blk, 0, stream>>>(nrm_bf, Wr_t,  br,    nullptr, retb,  Mc, Hc,     Dc);
    gemm_wmma_bf16<0, float><<<grid_for(3 * Hc),     blk, 0, stream>>>(nrm_bf, Wrl_t, brule, nullptr, rwb,   Mc, 3 * Hc, Dc);
    softmax3_kernel<<<(Mc * Hc + 255) / 256, blk, 0, stream>>>(rwb, Mc * Hc);
    gemm_wmma_bf16<2, float><<<grid_for(Dc),         blk, 0, stream>>>(nrm_bf, Wg_t,  bg,    nullptr, gateb, Mc, Dc,     Dc);

    // 4) sequential fast-weight scan (writes readout bf16 + final state)
    scan_kernel<<<Bc * Hc, blk, 0, stream>>>(qkvd, retb, rwb, state0, out_bf, state_f);

    // 5) output projection, gated residual
    gemm_wmma_bf16<0, float><<<grid_for(Dc), blk, 0, stream>>>(out_bf, Wo_t, nullptr, nullptr, attnb, Mc, Dc, INNERc);
    combine_kernel<<<2048, blk, 0, stream>>>(x, attnb, gateb, hb, Mc * Dc);

    // 6) FFN: LN -> W1+GELU (bf16 out) -> W2 + b2 + residual h -> d_out
    ln_bf16_kernel<<<Mc, 256, 0, stream>>>(hb, fnw, fnb, h2_bf);
    gemm_wmma_bf16<3, unsigned short><<<grid_for(HIDc), blk, 0, stream>>>(h2_bf, W1_t, b1, nullptr, ffn1_bf, Mc, HIDc, Dc);
    gemm_wmma_bf16<0, float><<<grid_for(Dc), blk, 0, stream>>>(ffn1_bf, W2_t, b2, hb, out_f, Mc, Dc, HIDc);
}
